// Decoder_8375186227967
// MI455X (gfx1250) — matrix operand order repaired and checked by reading, not compiled, not run on hardware
//
#include <hip/hip_runtime.h>
#include <hip/hip_bf16.h>
#include <math.h>

typedef __attribute__((ext_vector_type(16))) __bf16 v16bf;
typedef __attribute__((ext_vector_type(8)))  float  v8f;
typedef __attribute__((ext_vector_type(4)))  unsigned int tdm_v4u;
typedef __attribute__((ext_vector_type(8)))  int          tdm_v8i;
typedef __attribute__((ext_vector_type(4)))  int          tdm_v4i;

#define B_SZ   64
#define TMEL   1000
#define TENC   400
#define ENC    512
#define P1     128
#define P2     64
#define DEC    256
#define GN     1024   /* 4*DEC */
#define KG     832    /* P2 + ENC + DEC */
#define MEL    80
#define TILE_R 40     /* encoder rows per TDM tile  (10 tiles of 40 = 400) */
#define NTILE  10

/* ---------------- workspace layout (bytes, all 256-aligned) ---------------- */
#define O_PRE   ((size_t)0)                       /* bf16 [64000][64]   8,192,000  */
#define O_EBF   ((size_t)8192000)                 /* bf16 [64][400][512] 26,214,400 */
#define O_WQT   ((size_t)34406400)                /* bf16 [256][512]      262,144  */
#define O_WGT   ((size_t)34668544)                /* bf16 [832][1024]   1,703,936  */
#define O_WOT   ((size_t)36372480)                /* bf16 [256][80]        40,960  */
#define O_BG    ((size_t)36413440)                /* f32  [1024]            4,096  */
#define O_WP1   ((size_t)36417536)                /* bf16 wmma-B 3x8 tiles 24,576  */
#define O_WP2   ((size_t)36442112)                /* bf16 wmma-B 4x4 tiles 16,384  */

/* =========================================================================
 * Setup A: bf16 conversion + [n][k] -> [k][n] transposition of all weights
 * streamed by the recurrent kernel, plus combined gate bias and enc->bf16.
 * ========================================================================= */
__global__ __launch_bounds__(256) void setup_streams(
    const float* __restrict__ enc,
    const float* __restrict__ Wq,  const float* __restrict__ Wih,
    const float* __restrict__ Whh, const float* __restrict__ bih,
    const float* __restrict__ bhh, const float* __restrict__ Wout,
    __bf16* __restrict__ WqT, __bf16* __restrict__ WgT,
    __bf16* __restrict__ WoT, float* __restrict__ bg,
    __bf16* __restrict__ Ebf)
{
    size_t i = (size_t)blockIdx.x * 256 + threadIdx.x;
    if (i < 131072) {                       /* WqT[k][n] = Wq[n][k] */
        int k = (int)(i / 512), n = (int)(i % 512);
        WqT[i] = (__bf16)Wq[(size_t)n * 256 + k];
        return;
    }
    i -= 131072;
    if (i < 851968) {                       /* WgT[k][n], k<576 from Wih else Whh */
        int k = (int)(i / 1024), n = (int)(i % 1024);
        float v = (k < 576) ? Wih[(size_t)n * 576 + k]
                            : Whh[(size_t)n * 256 + (k - 576)];
        WgT[i] = (__bf16)v;
        return;
    }
    i -= 851968;
    if (i < 20480) {                        /* WoT[k][m] = Wout[m][k] */
        int k = (int)(i / 80), m = (int)(i % 80);
        WoT[i] = (__bf16)Wout[(size_t)m * 256 + k];
        return;
    }
    i -= 20480;
    if (i < 1024) { bg[i] = bih[i] + bhh[i]; return; }
    i -= 1024;
    if (i < (size_t)B_SZ * TENC * ENC) Ebf[i] = (__bf16)enc[i];
}

/* =========================================================================
 * Setup B: pack an [N][K] fp32 weight into a CDNA5 WMMA bf16 B-fragment
 * layout with contiguous k, matching this file's A side: tile(kt,nt) -> per lane l: N = nt*16 + (l&15),
 * element e -> K = kt*32 + 16*(l>>4) + e.  One 512-thread block per tile.
 * ========================================================================= */
__global__ __launch_bounds__(512) void setup_wmmaB(
    const float* __restrict__ src, __bf16* __restrict__ dst,
    int Ks, int Ns, int NT)
{
    int tile = blockIdx.x;
    int kt = tile / NT, nt = tile % NT;
    int lane = threadIdx.x & 31, e = threadIdx.x >> 5;
    int Kg = kt * 32 + ((lane >> 4) << 4) + e;
    int Ng = nt * 16 + (lane & 15);
    float v = (Kg < Ks && Ng < Ns) ? src[(size_t)Ng * Ks + Kg] : 0.0f;
    dst[((size_t)tile * 32 + lane) * 16 + e] = (__bf16)v;
}

/* =========================================================================
 * Prenet: fused Linear(80->128)+ReLU -> Linear(128->64)+ReLU via
 * V_WMMA_F32_16X16X32_BF16.  256 threads = 8 waves; each wave owns a
 * 16-row M-tile (500 blocks * 8 waves * 16 = 64000 rows exactly).
 * ========================================================================= */
__global__ __launch_bounds__(256) void prenet_kernel(
    const float*  __restrict__ mel,
    const __bf16* __restrict__ Wp1b, const float* __restrict__ bp1,
    const __bf16* __restrict__ Wp2b, const float* __restrict__ bp2,
    __bf16* __restrict__ pre)
{
    __shared__ float tile[8 * 16 * P1];            /* 64 KB */
    const int lane = threadIdx.x & 31;
    const int wave = threadIdx.x >> 5;
    const int m0   = (blockIdx.x * 8 + wave) * 16;
    const int mrow = m0 + (lane & 15);
    const int kh   = lane >> 4;
    float* myT = tile + wave * (16 * P1);

    /* A-fragment K mapping: K = 16*kh + e (matches packed B) */
    int kmap[16];
#pragma unroll
    for (int e = 0; e < 16; ++e) {
        kmap[e] = kh * 16 + e;
    }

    /* ---- layer 1: [16x80] x [80x128], K padded to 96 (3 k-tiles) ---- */
    v16bf a1[3];
#pragma unroll
    for (int kt = 0; kt < 3; ++kt)
#pragma unroll
        for (int e = 0; e < 16; ++e) {
            int kg = kt * 32 + kmap[e];
            a1[kt][e] = (__bf16)((kg < MEL) ? mel[(size_t)mrow * MEL + kg] : 0.0f);
        }
#pragma unroll
    for (int nt = 0; nt < 8; ++nt) {
        v8f acc = {};
#pragma unroll
        for (int kt = 0; kt < 3; ++kt) {
            v16bf bf = *(const v16bf*)(Wp1b + ((size_t)(kt * 8 + nt) * 32 + lane) * 16);
            acc = __builtin_amdgcn_wmma_f32_16x16x32_bf16(
                      false, a1[kt], false, bf, (short)0, acc, false, false);
        }
        float bias = bp1[nt * 16 + (lane & 15)];
#pragma unroll
        for (int i = 0; i < 8; ++i) {                   /* D: M = i+8*kh, N = lane&15 */
            float v = acc[i] + bias;
            myT[(i + 8 * kh) * P1 + nt * 16 + (lane & 15)] = v > 0.0f ? v : 0.0f;
        }
    }
    __syncthreads();

    /* ---- layer 2: [16x128] x [128x64], 4 k-tiles ---- */
    v16bf a2[4];
#pragma unroll
    for (int kt = 0; kt < 4; ++kt)
#pragma unroll
        for (int e = 0; e < 16; ++e)
            a2[kt][e] = (__bf16)myT[(lane & 15) * P1 + kt * 32 + kmap[e]];
#pragma unroll
    for (int nt = 0; nt < 4; ++nt) {
        v8f acc = {};
#pragma unroll
        for (int kt = 0; kt < 4; ++kt) {
            v16bf bf = *(const v16bf*)(Wp2b + ((size_t)(kt * 4 + nt) * 32 + lane) * 16);
            acc = __builtin_amdgcn_wmma_f32_16x16x32_bf16(
                      false, a2[kt], false, bf, (short)0, acc, false, false);
        }
        float bias = bp2[nt * 16 + (lane & 15)];
#pragma unroll
        for (int i = 0; i < 8; ++i) {
            float v = acc[i] + bias;
            v = v > 0.0f ? v : 0.0f;
            pre[(size_t)(m0 + i + 8 * kh) * P2 + nt * 16 + (lane & 15)] = (__bf16)v;
        }
    }
}

/* ------------------------- wave32 helpers ---------------------------------- */
__device__ __forceinline__ float wsum(float v) {
#pragma unroll
    for (int m = 16; m; m >>= 1) v += __shfl_xor(v, m, 32);
    return v;
}
__device__ __forceinline__ float sigm(float x) { return 1.0f / (1.0f + __expf(-x)); }

/* TDM: issue a 2-D tensor_load_to_lds of one E tile (TILE_R rows x 512 bf16).
 * D# packed per ISA 8.3-8.6: group0 = {count=1, lds_addr, global_addr, type=2},
 * group1 = {data_size=2B, tensor_dim0=512, tensor_dim1=TILE_R,
 *           tile_dim0=512, tile_dim1=TILE_R, tensor_dim0_stride=512}.
 * Uniform operands -> SGPR groups.  Tracked by TENSORcnt.
 * NOTE: this toolchain (clang-23 / therock-10.0 headers) exposes the 6-arg
 * builtin: (u32x4, i32x8, i32x4, i32x4, i32x8, i32 cpol). */
__device__ __forceinline__ void tdm_load_tile(unsigned int lds_addr,
                                              unsigned long long ga)
{
    tdm_v4u g0;
    g0[0] = 1u;                                   /* count=1, no gather        */
    g0[1] = lds_addr;                             /* D#.lds_addr               */
    g0[2] = (unsigned int)ga;                     /* global_addr[31:0]         */
    g0[3] = (unsigned int)(ga >> 32) | (2u << 30);/* global_addr[56:32]|type=2 */
    tdm_v8i g1;
    g1[0] = 0x10000;                              /* data_size=1 (2 bytes)     */
    g1[1] = (int)(512u << 16);                    /* tensor_dim0[15:0]=512     */
    g1[2] = (int)((unsigned)TILE_R << 16);        /* tensor_dim1[15:0]=TILE_R  */
    g1[3] = (int)(512u << 16);                    /* tile_dim0=512             */
    g1[4] = TILE_R;                               /* tile_dim1=TILE_R          */
    g1[5] = 512;                                  /* tensor_dim0_stride=512    */
    g1[6] = 0;
    g1[7] = 0;
    tdm_v4i z4 = {0, 0, 0, 0};
    tdm_v8i z8 = {0, 0, 0, 0, 0, 0, 0, 0};
    __builtin_amdgcn_tensor_load_to_lds(g0, g1, z4, z4, z8, 0);
}

/* =========================================================================
 * Recurrent decoder: one persistent workgroup per batch row (64 WGs),
 * 512 threads = 16 waves, 1000-step scan with LDS-resident h/c.
 * Attention is a single fused pass (online softmax + context accumulation)
 * over the encoder slice, which is TDM double-buffered into LDS in
 * TILE_R-row tiles (halves E traffic vs. separate energy/context passes).
 * ========================================================================= */
__global__ __launch_bounds__(512) void decoder_kernel(
    const __bf16* __restrict__ pre, const __bf16* __restrict__ Ebf,
    const __bf16* __restrict__ WqT, const __bf16* __restrict__ WgT,
    const __bf16* __restrict__ WoT,
    const float* __restrict__ bq,  const float* __restrict__ we,
    const float* __restrict__ be,  const float* __restrict__ bg,
    const float* __restrict__ bout, float* __restrict__ out)
{
    __shared__ __bf16 sh_E[2][TILE_R * ENC];      /* 80 KB TDM double buffer */
    __shared__ float  sh_ctxw[16][ENC];           /* 32 KB per-wave partials */
    __shared__ float  sh_h[DEC], sh_c[DEC], sh_q[ENC];
    __shared__ float  sh_act[KG], sh_gates[GN], sh_we[ENC];
    __shared__ float  sh_m[16], sh_s[16];

    const int tid  = threadIdx.x;
    const int lane = tid & 31, wave = tid >> 5;
    const int b    = blockIdx.x;
    const __bf16* Erow = Ebf + (size_t)b * TENC * ENC;
    const float be0 = be[0];
    const unsigned int ldsE0 = (unsigned int)(size_t)&sh_E[0][0];
    const unsigned int ldsE1 = (unsigned int)(size_t)&sh_E[1][0];

    if (tid < DEC) { sh_h[tid] = 0.0f; sh_c[tid] = 0.0f; }
    sh_we[tid] = we[tid];
    __syncthreads();

    for (int t = 0; t < TMEL; ++t) {
        /* kick off TDM for tile 0 of this step (buffers free: end barrier) */
        if (tid == 0)
            tdm_load_tile(ldsE0, (unsigned long long)(size_t)Erow);

        /* x_t -> sh_act[0..63]; prefetch next step's row into L2 */
        if (tid < P2) {
            sh_act[tid] = (float)pre[((size_t)b * TMEL + t) * P2 + tid];
            if (t + 1 < TMEL)
                __builtin_prefetch(&pre[((size_t)b * TMEL + t + 1) * P2 + tid], 0, 2);
        }
        /* q = h @ Wq^T + bq  (WqT is [k][n] bf16 -> lane-coalesced) */
        {
            float acc = bq[tid];
#pragma unroll 4
            for (int k = 0; k < DEC; ++k)
                acc += sh_h[k] * (float)WqT[(size_t)k * ENC + tid];
            sh_q[tid] = acc;
        }
        __syncthreads();

        /* ---- fused attention: online softmax + context, tile by tile ---- */
        float m_w = -3.4e38f, s_w = 0.0f;
        float cacc[16];
#pragma unroll
        for (int i = 0; i < 16; ++i) cacc[i] = 0.0f;

        for (int tile = 0; tile < NTILE; ++tile) {
            const int cur = tile & 1;
            if (tid == 0) __builtin_amdgcn_s_wait_tensorcnt(0);
            __syncthreads();                       /* tile `cur` ready in LDS */
            if (tid == 0 && tile + 1 < NTILE)
                tdm_load_tile(cur ? ldsE0 : ldsE1,
                              (unsigned long long)(size_t)
                                  (Erow + (size_t)(tile + 1) * TILE_R * ENC));

            const __bf16* Et = &sh_E[cur][0];
            for (int r = wave; r < TILE_R; r += 16) {
                const __bf16* er = Et + r * ENC;
                float ev[16], p = 0.0f;
#pragma unroll
                for (int i = 0; i < 16; ++i) {
                    int j = lane + 32 * i;
                    float e = (float)er[j];
                    ev[i] = e;
                    p += tanhf(sh_q[j] + e) * sh_we[j];
                }
                float et = wsum(p) + be0;          /* wave-uniform energy */
                if (et > m_w) {                    /* uniform branch */
                    float sc = __expf(m_w - et);
                    s_w = s_w * sc + 1.0f;
#pragma unroll
                    for (int i = 0; i < 16; ++i) cacc[i] = cacc[i] * sc + ev[i];
                    m_w = et;
                } else {
                    float wgt = __expf(et - m_w);
                    s_w += wgt;
#pragma unroll
                    for (int i = 0; i < 16; ++i) cacc[i] += wgt * ev[i];
                }
            }
            __syncthreads();                       /* done reading tile `cur` */
        }

        /* merge the 16 per-wave (m, s, ctx) partials */
        if (lane == 0) { sh_m[wave] = m_w; sh_s[wave] = s_w; }
#pragma unroll
        for (int i = 0; i < 16; ++i) sh_ctxw[wave][lane + 32 * i] = cacc[i];
        __syncthreads();
        {
            float M = -3.4e38f;
#pragma unroll
            for (int w = 0; w < 16; ++w) M = fmaxf(M, sh_m[w]);
            float S = 0.0f, acc = 0.0f;
#pragma unroll
            for (int w = 0; w < 16; ++w) {
                float sc = __expf(sh_m[w] - M);
                S   += sh_s[w] * sc;
                acc += sh_ctxw[w][tid] * sc;
            }
            sh_act[P2 + tid] = acc / S;            /* context[tid] */
        }
        if (tid < DEC) sh_act[P2 + ENC + tid] = sh_h[tid];
        __syncthreads();

        /* gates = [x|ctx|h] @ WgT + (b_ih+b_hh); 2 outputs per thread */
        {
            float a0 = bg[tid], a1 = bg[tid + 512];
#pragma unroll 4
            for (int k = 0; k < KG; ++k) {
                float a = sh_act[k];
                a0 += a * (float)WgT[(size_t)k * GN + tid];
                a1 += a * (float)WgT[(size_t)k * GN + tid + 512];
            }
            sh_gates[tid]       = a0;
            sh_gates[tid + 512] = a1;
        }
        __syncthreads();

        /* LSTM cell (torch order i,f,g,o) */
        if (tid < DEC) {
            float ig = sigm(sh_gates[tid]);
            float fg = sigm(sh_gates[DEC + tid]);
            float gg = tanhf(sh_gates[2 * DEC + tid]);
            float og = sigm(sh_gates[3 * DEC + tid]);
            float cn = fg * sh_c[tid] + ig * gg;
            sh_c[tid] = cn;
            sh_h[tid] = og * tanhf(cn);
        }
        __syncthreads();

        /* mel_out = h_new @ Wout^T + b_out */
        if (tid < MEL) {
            float acc = bout[tid];
#pragma unroll 4
            for (int k = 0; k < DEC; ++k)
                acc += sh_h[k] * (float)WoT[(size_t)k * MEL + tid];
            out[((size_t)b * TMEL + t) * MEL + tid] = acc;
        }
        __syncthreads();
    }
}

/* ========================================================================= */
extern "C" void kernel_launch(void* const* d_in, const int* in_sizes, int n_in,
                              void* d_out, int out_size, void* d_ws, size_t ws_size,
                              hipStream_t stream)
{
    (void)in_sizes; (void)n_in; (void)out_size; (void)ws_size;
    const float* mel  = (const float*)d_in[0];
    const float* enc  = (const float*)d_in[1];
    const float* Wp1  = (const float*)d_in[2];
    const float* bp1  = (const float*)d_in[3];
    const float* Wp2  = (const float*)d_in[4];
    const float* bp2  = (const float*)d_in[5];
    const float* Wq   = (const float*)d_in[6];
    const float* bq   = (const float*)d_in[7];
    const float* We   = (const float*)d_in[8];
    const float* be   = (const float*)d_in[9];
    const float* Wih  = (const float*)d_in[10];
    const float* Whh  = (const float*)d_in[11];
    const float* bih  = (const float*)d_in[12];
    const float* bhh  = (const float*)d_in[13];
    const float* Wout = (const float*)d_in[14];
    const float* bout = (const float*)d_in[15];

    char* ws = (char*)d_ws;
    __bf16* pre  = (__bf16*)(ws + O_PRE);
    __bf16* Ebf  = (__bf16*)(ws + O_EBF);
    __bf16* WqT  = (__bf16*)(ws + O_WQT);
    __bf16* WgT  = (__bf16*)(ws + O_WGT);
    __bf16* WoT  = (__bf16*)(ws + O_WOT);
    float*  bg   = (float*) (ws + O_BG);
    __bf16* Wp1b = (__bf16*)(ws + O_WP1);
    __bf16* Wp2b = (__bf16*)(ws + O_WP2);

    /* one-time operand prep */
    size_t nA = 131072 + 851968 + 20480 + 1024 + (size_t)B_SZ * TENC * ENC;
    setup_streams<<<(unsigned)((nA + 255) / 256), 256, 0, stream>>>(
        enc, Wq, Wih, Whh, bih, bhh, Wout, WqT, WgT, WoT, bg, Ebf);
    setup_wmmaB<<<3 * 8, 512, 0, stream>>>(Wp1, Wp1b, MEL, P1, 8);
    setup_wmmaB<<<4 * 4, 512, 0, stream>>>(Wp2, Wp2b, P1, P2, 4);

    /* WMMA prenet over all 64000 rows */
    prenet_kernel<<<500, 256, 0, stream>>>(mel, Wp1b, bp1, Wp2b, bp2, pre);

    /* persistent per-batch-row recurrent decoder */
    decoder_kernel<<<B_SZ, 512, 0, stream>>>(
        pre, Ebf, WqT, WgT, WoT, bq, We, be, bg, bout, (float*)d_out);
}
